// recurrent_GConvGRU_54202487275563
// MI455X (gfx1250) — matrix operand order repaired and checked by reading, not compiled, not run on hardware
//
#include <hip/hip_runtime.h>
#include <stdint.h>

typedef __attribute__((ext_vector_type(16))) _Float16 v16h;
typedef __attribute__((ext_vector_type(8)))  _Float16 v8h;
typedef __attribute__((ext_vector_type(4)))  _Float16 v4h;
typedef __attribute__((ext_vector_type(8)))  float    v8f;
typedef __attribute__((ext_vector_type(4)))  float    v4f;

#define PADN  3000
#define MROWS 2800
#define NF    8192
#define H1    256
#define OUTF  128
#define NEDGE 89600
#define MPAD  3008   // gate GEMM M padded to 188*16
#define KCAT  512    // concat K: [h_in | tx1]

// ---------------- utility kernels ----------------

__global__ void k_zero_u32(uint32_t* __restrict__ p, int n) {
    int i = blockIdx.x * blockDim.x + threadIdx.x;
    if (i < n) p[i] = 0u;
}

__global__ void k_cast_f32_f16_v4(const float* __restrict__ src,
                                  _Float16* __restrict__ dst, int n4) {
    int i = blockIdx.x * blockDim.x + threadIdx.x;
    if (i < n4) {
        v4f v = ((const v4f*)src)[i];
        v4h h;
        h.x = (_Float16)v.x; h.y = (_Float16)v.y;
        h.z = (_Float16)v.z; h.w = (_Float16)v.w;
        ((v4h*)dst)[i] = h;
    }
}

// wt[n*K + k] = (f16) w[k*N + n]
__global__ void k_transpose_cast(const float* __restrict__ w,
                                 _Float16* __restrict__ wt, int K, int N) {
    int i = blockIdx.x * blockDim.x + threadIdx.x;
    if (i < K * N) {
        int k = i / N, n = i % N;
        wt[n * K + k] = (_Float16)w[i];
    }
}

// Bt2[g2][n][k]: g2=0 -> gate0 (Z), g2=1 -> gate2 (H~); k<256 from Wx0, else Wx1
__global__ void k_build_bt2(const float* __restrict__ Wx0,
                            const float* __restrict__ Wx1,
                            _Float16* __restrict__ bt2) {
    int i = blockIdx.x * blockDim.x + threadIdx.x;
    if (i >= 2 * OUTF * KCAT) return;
    int g2  = i / (OUTF * KCAT);
    int rem = i % (OUTF * KCAT);
    int n = rem / KCAT, k = rem % KCAT;
    int g = g2 * 2;  // gates 0 and 2
    float v = (k < H1) ? Wx0[(g * H1 + k) * OUTF + n]
                       : Wx1[(g * H1 + (k - H1)) * OUTF + n];
    bt2[i] = (_Float16)v;
}

// ---------------- graph weights ----------------

__global__ void k_deg(const long long* __restrict__ ei, float* __restrict__ deg) {
    int e = blockIdx.x * blockDim.x + threadIdx.x;
    if (e < NEDGE) {
        int r = (int)ei[e];
        int c = (int)ei[NEDGE + e];
        if (r != c) atomicAdd(&deg[r], 1.0f);
    }
}

__global__ void k_dis(float* __restrict__ deg) {
    int i = blockIdx.x * blockDim.x + threadIdx.x;
    if (i < PADN) {
        float d = deg[i];
        deg[i] = (d > 0.0f) ? (1.0f / sqrtf(d)) : 0.0f;
    }
}

__global__ void k_edgew(const long long* __restrict__ ei,
                        const float* __restrict__ dis, float* __restrict__ w) {
    int e = blockIdx.x * blockDim.x + threadIdx.x;
    if (e < NEDGE) {
        int r = (int)ei[e];
        int c = (int)ei[NEDGE + e];
        w[e] = (r != c) ? (-dis[r] * dis[c]) : 0.0f;
    }
}

// ---------------- WMMA helpers ----------------

// A 16x32 f16 layout: lane L (L<16): a[0..15]=K[koff..+15], koff=0
// lane L>=16: koff=16.
static __device__ inline v16h load_a_tile(const _Float16* __restrict__ rowp,
                                          int k0, int koff) {
    v8h lo = *(const v8h*)(rowp + k0 + koff);
    v8h hi = *(const v8h*)(rowp + k0 + 8 + koff);
    return __builtin_shufflevector(lo, hi, 0, 1, 2, 3, 4, 5, 6, 7,
                                   8, 9, 10, 11, 12, 13, 14, 15);
}

static __device__ inline v8f wmma_f16(v16h a, v16h b, v8f c) {
    return __builtin_amdgcn_wmma_f32_16x16x32_f16(false, a, false, b,
                                                  (short)0, c, false, false);
}

// 16B async copy global -> LDS (own counter: ASYNCcnt; overlaps with VALU/WMMA)
static __device__ inline void async_copy16(uint32_t lds_off, const void* gaddr) {
    asm volatile("global_load_async_to_lds_b128 %0, %1, off"
                 :: "v"(lds_off), "v"(gaddr) : "memory");
}

// ---------------- GEMM1: h_in = relu(x @ fc0_w + b) -> Acat[:, 0:256] as f16 ---
// grid = (175, 4), block = 32 (one wave). Wave computes a 16x64 tile.
// Double-buffered async global->LDS staging of K=64 slabs:
//   sA slab: 16 rows x 64 halves (2KB, 128 x 16B chunks, 4 async ops)
//   sB slab: 64 cols x 64 halves (8KB, 512 x 16B chunks, 16 async ops)
// Issue next slab (20 async ops) -> s_wait_asynccnt 20 (current slab done,
// async loads complete in order) -> 8 WMMAs fed by ds_load_b128.
__global__ void k_gemm1(const _Float16* __restrict__ A,   // [2800][8192]
                        const _Float16* __restrict__ Bt,  // [256][8192] (transposed)
                        const float* __restrict__ bias,   // fc0_b [256]
                        _Float16* __restrict__ Acat) {    // [3008][512]
    __shared__ _Float16 sA[2][16 * 64];
    __shared__ _Float16 sB[2][64 * 64];

    int lane = threadIdx.x;
    int m0 = blockIdx.x * 16;
    int n0 = blockIdx.y * 64;
    int koff = (lane >> 4) * 16;   // A half-row split
    int kgrp = (lane >> 4) * 16;   // B K-group split

    // issue one K=64 slab into buffer bsel, starting at global k0
    auto issue_slab = [&](int bsel, int k0) {
#pragma unroll
        for (int i = 0; i < 4; ++i) {            // A: 128 chunks of 16B
            int c = i * 32 + lane;
            int row = c >> 3, sub = c & 7;       // 8 chunks per 128B row
            const void* g = (const void*)(A + (size_t)(m0 + row) * NF + k0 + sub * 8);
            uint32_t l = (uint32_t)(uintptr_t)&sA[bsel][c * 8];
            async_copy16(l, g);
        }
#pragma unroll
        for (int i = 0; i < 16; ++i) {           // B: 512 chunks of 16B
            int c = i * 32 + lane;
            int col = c >> 3, sub = c & 7;       // 8 chunks per 128B col
            const void* g = (const void*)(Bt + (size_t)(n0 + col) * NF + k0 + sub * 8);
            uint32_t l = (uint32_t)(uintptr_t)&sB[bsel][c * 8];
            async_copy16(l, g);
        }
    };

    v8f acc0 = {}, acc1 = {}, acc2 = {}, acc3 = {};
    issue_slab(0, 0);
    int buf = 0;
    for (int k0 = 0; k0 < NF; k0 += 64) {
        int k1 = k0 + 64;
        if (k1 >= NF) k1 = 0;                    // clamped prefetch (unused result)
        issue_slab(buf ^ 1, k1);
        // wait for current slab only; the 20 just-issued ops keep streaming
        asm volatile("s_wait_asynccnt 20" ::: "memory");
#pragma unroll
        for (int s = 0; s < 2; ++s) {            // two K=32 WMMA steps per slab
            int ar = (lane & 15) * 64 + s * 32;
            v8h alo = *(const v8h*)&sA[buf][ar + koff];
            v8h ahi = *(const v8h*)&sA[buf][ar + 8 + koff];
            v16h a = __builtin_shufflevector(alo, ahi, 0, 1, 2, 3, 4, 5, 6, 7,
                                             8, 9, 10, 11, 12, 13, 14, 15);
            v16h b0 = *(const v16h*)&sB[buf][(0 * 16 + (lane & 15)) * 64 + s * 32 + kgrp];
            v16h b1 = *(const v16h*)&sB[buf][(1 * 16 + (lane & 15)) * 64 + s * 32 + kgrp];
            v16h b2 = *(const v16h*)&sB[buf][(2 * 16 + (lane & 15)) * 64 + s * 32 + kgrp];
            v16h b3 = *(const v16h*)&sB[buf][(3 * 16 + (lane & 15)) * 64 + s * 32 + kgrp];
            acc0 = wmma_f16(a, b0, acc0);
            acc1 = wmma_f16(a, b1, acc1);
            acc2 = wmma_f16(a, b2, acc2);
            acc3 = wmma_f16(a, b3, acc3);
        }
        buf ^= 1;
    }

    int rbase = m0 + (lane >> 4) * 8;
    v8f accs[4] = {acc0, acc1, acc2, acc3};
#pragma unroll
    for (int t = 0; t < 4; ++t) {
        int col = n0 + t * 16 + (lane & 15);
        float bv = bias[col];
#pragma unroll
        for (int v = 0; v < 8; ++v) {
            float val = accs[t][v] + bv;
            val = val > 0.0f ? val : 0.0f;        // relu
            Acat[(size_t)(rbase + v) * KCAT + col] = (_Float16)val;
        }
    }
}

// ---------------- scatter: tx1[col] += h_in[row] * w[e] ----------------
__global__ void k_scatter(const long long* __restrict__ ei,
                          const float* __restrict__ w,
                          const _Float16* __restrict__ Acat,  // h_in in cols 0..255
                          float* __restrict__ tx1) {
    int e = blockIdx.x;
    int f = threadIdx.x;
    float we = w[e];
    if (we == 0.0f) return;
    int r = (int)ei[e];
    int c = (int)ei[NEDGE + e];
    float hv = (float)Acat[(size_t)r * KCAT + f];
    if (hv != 0.0f) atomicAdd(&tx1[(size_t)c * H1 + f], hv * we);
}

// tx1 f32 -> Acat[:, 256:512] f16
__global__ void k_cast_tx1(const float* __restrict__ tx1,
                           _Float16* __restrict__ Acat) {
    int i = blockIdx.x * blockDim.x + threadIdx.x;
    if (i < PADN * H1) {
        int r = i / H1, f = i % H1;
        Acat[(size_t)r * KCAT + H1 + f] = (_Float16)tx1[i];
    }
}

// ---------------- fused gate GEMM + GRU epilogue ----------------
// H = (1 - sigmoid(Sz)) * tanh(Sh);  Sg = Acat @ Btg + bx[g] + bh[g]
struct Stage2 { v16h a, b0, b1; };

__global__ void k_gates(const _Float16* __restrict__ Acat, // [3008][512]
                        const _Float16* __restrict__ Bt2,  // [2][128][512]
                        const float* __restrict__ bx,      // [3][128]
                        const float* __restrict__ bh,      // [3][128]
                        float* __restrict__ Hout) {        // [3000][128]
    int lane = threadIdx.x;
    int m0 = blockIdx.x * 16;
    int n0 = blockIdx.y * 16;
    int koff = (lane >> 4) * 16;
    int kgrp = (lane >> 4) * 16;
    const _Float16* arow = Acat + (size_t)(m0 + (lane & 15)) * KCAT;
    const _Float16* bz   = Bt2 + (size_t)(n0 + (lane & 15)) * KCAT + kgrp;
    const _Float16* bhp  = bz + (size_t)OUTF * KCAT;

    v8f accz = {}, acch = {};
    Stage2 cur;
    cur.a  = load_a_tile(arow, 0, koff);
    cur.b0 = *(const v16h*)(bz + 0);
    cur.b1 = *(const v16h*)(bhp + 0);
#pragma unroll 4
    for (int k0 = 0; k0 < KCAT; k0 += 32) {
        int k1 = k0 + 32;
        if (k1 >= KCAT) k1 = 0;                   // clamped prefetch (unused result)
        Stage2 nxt;
        nxt.a  = load_a_tile(arow, k1, koff);
        nxt.b0 = *(const v16h*)(bz + k1);
        nxt.b1 = *(const v16h*)(bhp + k1);
        accz = wmma_f16(cur.a, cur.b0, accz);
        acch = wmma_f16(cur.a, cur.b1, acch);
        cur = nxt;
    }

    int col = n0 + (lane & 15);
    float bzv = bx[col] + bh[col];                       // gate 0 (Z)
    float bhv = bx[2 * OUTF + col] + bh[2 * OUTF + col]; // gate 2 (H~)
    int rbase = m0 + (lane >> 4) * 8;
#pragma unroll
    for (int v = 0; v < 8; ++v) {
        int r = rbase + v;
        if (r < PADN) {
            float z  = 1.0f / (1.0f + __expf(-(accz[v] + bzv)));
            float ht = tanhf(acch[v] + bhv);
            Hout[(size_t)r * OUTF + col] = (1.0f - z) * ht;  // H0==0 => Z*H0 drops
        }
    }
}

// out[r] = relu(H[r]) . fc_w + fc_b
__global__ void k_out(const float* __restrict__ H, const float* __restrict__ fcw,
                      const float* __restrict__ fcb, float* __restrict__ out) {
    int r = blockIdx.x * blockDim.x + threadIdx.x;
    if (r < MROWS) {
        float s = 0.0f;
#pragma unroll 4
        for (int n = 0; n < OUTF; ++n) {
            float h = H[(size_t)r * OUTF + n];
            s += (h > 0.0f ? h : 0.0f) * fcw[n];
        }
        out[r] = s + fcb[0];
    }
}

// ---------------- host launcher ----------------

extern "C" void kernel_launch(void* const* d_in, const int* in_sizes, int n_in,
                              void* d_out, int out_size, void* d_ws, size_t ws_size,
                              hipStream_t stream) {
    const float*     x     = (const float*)d_in[0];
    const long long* ei    = (const long long*)d_in[1];  // int64 [2][89600]
    const float*     fc0_w = (const float*)d_in[2];
    const float*     fc0_b = (const float*)d_in[3];
    const float*     Wx0   = (const float*)d_in[4];
    const float*     Wx1   = (const float*)d_in[5];
    const float*     bx    = (const float*)d_in[6];
    // d_in[7] (Wh0), d_in[8] (Wh1) multiply H0==0 -> unused
    const float*     bh    = (const float*)d_in[9];
    const float*     fc_w  = (const float*)d_in[10];
    const float*     fc_b  = (const float*)d_in[11];

    float* out  = (float*)d_out;          // [2800]
    float* Hout = out + MROWS;            // [3000*128]

    // workspace carve-up (256B aligned)
    uintptr_t base = (uintptr_t)d_ws;
    size_t off = 0;
    auto carve = [&](size_t bytes) -> void* {
        off = (off + 255) & ~(size_t)255;
        void* p = (void*)(base + off);
        off += bytes;
        return p;
    };
    _Float16* x_h  = (_Float16*)carve((size_t)MROWS * NF * 2);   // 45.9 MB
    _Float16* w0t  = (_Float16*)carve((size_t)H1 * NF * 2);      //  4.2 MB
    _Float16* Acat = (_Float16*)carve((size_t)MPAD * KCAT * 2);  //  3.1 MB
    _Float16* Bt2  = (_Float16*)carve((size_t)2 * OUTF * KCAT * 2);
    float*    tx1  = (float*)carve((size_t)PADN * H1 * 4);       //  3.1 MB
    float*    deg  = (float*)carve((size_t)PADN * 4);
    float*    wbuf = (float*)carve((size_t)NEDGE * 4);
    (void)ws_size; (void)in_sizes; (void)n_in; (void)out_size;

    const int T = 256;
    // zero accumulators / padded buffers
    k_zero_u32<<<(MPAD * KCAT / 2 + T - 1) / T, T, 0, stream>>>((uint32_t*)Acat, MPAD * KCAT / 2);
    k_zero_u32<<<(PADN * H1 + T - 1) / T, T, 0, stream>>>((uint32_t*)tx1, PADN * H1);
    k_zero_u32<<<(PADN + T - 1) / T, T, 0, stream>>>((uint32_t*)deg, PADN);

    // precision casts / operand layouts
    int n4 = MROWS * NF / 4;
    k_cast_f32_f16_v4<<<(n4 + T - 1) / T, T, 0, stream>>>(x, x_h, n4);
    k_transpose_cast<<<(NF * H1 + T - 1) / T, T, 0, stream>>>(fc0_w, w0t, NF, H1);
    k_build_bt2<<<(2 * OUTF * KCAT + T - 1) / T, T, 0, stream>>>(Wx0, Wx1, Bt2);

    // Chebyshev edge weights
    k_deg<<<(NEDGE + T - 1) / T, T, 0, stream>>>(ei, deg);
    k_dis<<<(PADN + T - 1) / T, T, 0, stream>>>(deg);
    k_edgew<<<(NEDGE + T - 1) / T, T, 0, stream>>>(ei, deg, wbuf);

    // h_in = relu(x @ fc0_w + b)  (WMMA f16 -> f32, async-LDS pipelined)
    k_gemm1<<<dim3(MROWS / 16, H1 / 64), 32, 0, stream>>>(x_h, w0t, fc0_b, Acat);

    // tx1 = scatter_add over edges
    k_scatter<<<NEDGE, H1, 0, stream>>>(ei, wbuf, Acat, tx1);
    k_cast_tx1<<<(PADN * H1 + T - 1) / T, T, 0, stream>>>(tx1, Acat);

    // fused gate GEMMs + GRU update -> H
    k_gates<<<dim3(MPAD / 16, OUTF / 16), 32, 0, stream>>>(Acat, Bt2, bx, bh, Hout);

    // final projection
    k_out<<<(MROWS + T - 1) / T, T, 0, stream>>>(Hout, fc_w, fc_b, out);
}
